// CrossAttention_72275709657317
// MI455X (gfx1250) — matrix operand order repaired and checked by reading, not compiled, not run on hardware
//
#include <hip/hip_runtime.h>

// ---------------------------------------------------------------------------
// Cross-attention with softmax over the QUERY axis (faithful quirk):
//   Q = x Wq^T + bq            [B,S,D]
//   K = y Wk^T + bk            [B,S,D]
//   V = (y WvR^T + bvR) WvL^T + bvL   [B,S,E]  (stored transposed: Vt[E,M])
//   s = Q K^T / sqrt(D);  attn = softmax over q (per (b,k) column)
//   out = attn @ V             [B,S,E]
// B=4, S=2048, E=1024, D=64. All GEMMs on v_wmma_f32_16x16x32_bf16.
// Weights pre-converted to bf16 once; fragment loads grouped ahead of WMMA
// groups so one clause/wait covers several matrix ops.
// ---------------------------------------------------------------------------

#define B_ 4
#define S_ 2048
#define E_ 1024
#define D_ 64
#define M_ (B_ * S_)   // 8192 total rows

typedef __attribute__((ext_vector_type(16))) __bf16 v16bf;
typedef __attribute__((ext_vector_type(8)))  __bf16 v8bf;
typedef __attribute__((ext_vector_type(8)))  float  v8f;
typedef __attribute__((ext_vector_type(4)))  float  v4f;

static __device__ __forceinline__ __bf16 f2bf(float f) {
  return static_cast<__bf16>(f);   // hardware RNE cvt (v_cvt_pk_bf16_f32)
}

static __device__ __forceinline__ v8f wmma_bf16(v16bf a, v16bf b, v8f c) {
  return __builtin_amdgcn_wmma_f32_16x16x32_bf16(
      /*neg_a=*/false, a, /*neg_b=*/false, b,
      /*c_mod=*/(short)0, c, /*reuse_a=*/false, /*reuse_b=*/false);
}

static __device__ __forceinline__ v16bf cat8(v8bf lo, v8bf hi) {
  return __builtin_shufflevector(lo, hi, 0, 1, 2, 3, 4, 5, 6, 7,
                                 8, 9, 10, 11, 12, 13, 14, 15);
}

// A-matrix 16x32 fragment (bf16) from f32 row-major source, converting.
// ISA layout: lanes 0-15 row=lane, K {0..7,16..23}; lanes 16-31 row=lane-16,
// K {8..15, 24..31}.
static __device__ __forceinline__ v16bf afrag_f32(const float* p, int ld,
                                                  size_t rowBase, int k0, int lane) {
  size_t row = rowBase + (size_t)(lane & 15);
  int koff = (lane < 16) ? 0 : 8;
  const float* r0 = p + row * (size_t)ld + k0 + koff;
  v4f a0 = *(const v4f*)(r0);
  v4f a1 = *(const v4f*)(r0 + 4);
  v4f a2 = *(const v4f*)(r0 + 16);
  v4f a3 = *(const v4f*)(r0 + 20);
  v16bf a;
#pragma unroll
  for (int j = 0; j < 4; ++j) {
    a[j]      = f2bf(a0[j]);
    a[4 + j]  = f2bf(a1[j]);
    a[8 + j]  = f2bf(a2[j]);
    a[12 + j] = f2bf(a3[j]);
  }
  return a;
}

// Same A layout from a bf16 row-major source (vectorized 16B loads).
static __device__ __forceinline__ v16bf afrag_bf16(const __bf16* p, int ld,
                                                   size_t rowBase, int k0, int lane) {
  size_t row = rowBase + (size_t)(lane & 15);
  int koff = (lane < 16) ? 0 : 8;
  const __bf16* r0 = p + row * (size_t)ld + k0 + koff;
  v8bf lo = *(const v8bf*)(r0);
  v8bf hi = *(const v8bf*)(r0 + 16);
  return cat8(lo, hi);
}

// B-matrix 32x16 fragment where logical B[k][n] = W[n][k] (W row-major in k,
// bf16): two 16B loads per lane. ISA layout: lanes 0-15 col=lane K {0..7,16..23},
// lanes 16-31 col=lane-16 K {8..15,24..31}.
static __device__ __forceinline__ v16bf bfrag_wT_bf16(const __bf16* Mrows, int ld,
                                                      size_t colRowBase, int k0, int lane) {
  size_t row = colRowBase + (size_t)(lane & 15);
  int koff = (lane < 16) ? 0 : 8;
  const __bf16* r0 = Mrows + row * (size_t)ld + k0 + koff;
  v8bf lo = *(const v8bf*)(r0);
  v8bf hi = *(const v8bf*)(r0 + 16);
  return cat8(lo, hi);
}

// ---------------------------------------------------------------------------
// Kernel W: one-shot f32 -> bf16 conversion of the four weight matrices
// (each 64K elements). 64 blocks x 256 threads, 4 elements per thread.
// ---------------------------------------------------------------------------
__global__ void k_cvt_w(const float* __restrict__ Wq, const float* __restrict__ Wk,
                        const float* __restrict__ WvR, const float* __restrict__ WvL,
                        __bf16* __restrict__ Wqb, __bf16* __restrict__ Wkb,
                        __bf16* __restrict__ WvRb, __bf16* __restrict__ WvLb) {
  int i = (blockIdx.x * 256 + threadIdx.x) * 4;   // 0 .. 65532
  v4f q = *(const v4f*)(Wq + i);
  v4f k = *(const v4f*)(Wk + i);
  v4f r = *(const v4f*)(WvR + i);
  v4f l = *(const v4f*)(WvL + i);
#pragma unroll
  for (int j = 0; j < 4; ++j) {
    Wqb[i + j]  = f2bf(q[j]);
    Wkb[i + j]  = f2bf(k[j]);
    WvRb[i + j] = f2bf(r[j]);
    WvLb[i + j] = f2bf(l[j]);
  }
}

// ---------------------------------------------------------------------------
// Kernel A: Q = x Wq^T + bq ; K = y Wk^T + bk  -> bf16 [M, 64]
// One wave per 16-row tile. 128 blocks x 128 threads (4 waves).
// All 8 B-fragments loaded as one group, then 8 WMMAs back-to-back.
// ---------------------------------------------------------------------------
__global__ void k_proj_qk(const float* __restrict__ x, const float* __restrict__ y,
                          const __bf16* __restrict__ Wqb, const float* __restrict__ bq,
                          const __bf16* __restrict__ Wkb, const float* __restrict__ bk,
                          __bf16* __restrict__ Qb, __bf16* __restrict__ Kb) {
  int lane = threadIdx.x & 31;
  int wave = threadIdx.x >> 5;
  size_t rowBase = ((size_t)blockIdx.x * 4 + wave) * 16;

  v8f accq[4] = {}, acck[4] = {};
  for (int kc = 0; kc < E_; kc += 32) {
    v16bf ax = afrag_f32(x, E_, rowBase, kc, lane);
    v16bf ay = afrag_f32(y, E_, rowBase, kc, lane);
    v16bf bwq[4], bwk[4];
#pragma unroll
    for (int nt = 0; nt < 4; ++nt) {
      bwq[nt] = bfrag_wT_bf16(Wqb, E_, (size_t)(nt * 16), kc, lane);
      bwk[nt] = bfrag_wT_bf16(Wkb, E_, (size_t)(nt * 16), kc, lane);
    }
#pragma unroll
    for (int nt = 0; nt < 4; ++nt) {
      accq[nt] = wmma_bf16(ax, bwq[nt], accq[nt]);
      acck[nt] = wmma_bf16(ay, bwk[nt], acck[nt]);
    }
  }
  int col = lane & 15;
  int rOff = (lane < 16) ? 0 : 8;
#pragma unroll
  for (int nt = 0; nt < 4; ++nt) {
#pragma unroll
    for (int r = 0; r < 8; ++r) {
      size_t m = rowBase + rOff + r;
      Qb[m * D_ + nt * 16 + col] = f2bf(accq[nt][r] + bq[nt * 16 + col]);
      Kb[m * D_ + nt * 16 + col] = f2bf(acck[nt][r] + bk[nt * 16 + col]);
    }
  }
}

// ---------------------------------------------------------------------------
// Kernel B: V = (y WvR^T + bvR) WvL^T + bvL -> bf16 TRANSPOSED: Vt[E, M]
// C-layout -> A-layout transpose of the low-rank P tile via LDS.
// Transposed V store: C layout (lane=col, vgpr=row) -> 8 contiguous bf16/lane.
// ---------------------------------------------------------------------------
__global__ void k_proj_v(const float* __restrict__ y,
                         const __bf16* __restrict__ WvRb, const float* __restrict__ bvR,
                         const __bf16* __restrict__ WvLb, const float* __restrict__ bvL,
                         __bf16* __restrict__ Vt) {
  __shared__ __bf16 ldsP[4][16 * 64];
  int lane = threadIdx.x & 31;
  int wave = threadIdx.x >> 5;
  size_t rowBase = ((size_t)blockIdx.x * 4 + wave) * 16;

  v8f accp[4] = {};
  for (int kc = 0; kc < E_; kc += 32) {
    v16bf ay = afrag_f32(y, E_, rowBase, kc, lane);
    v16bf bw[4];
#pragma unroll
    for (int nt = 0; nt < 4; ++nt)
      bw[nt] = bfrag_wT_bf16(WvRb, E_, (size_t)(nt * 16), kc, lane);
#pragma unroll
    for (int nt = 0; nt < 4; ++nt)
      accp[nt] = wmma_bf16(ay, bw[nt], accp[nt]);
  }
  int col = lane & 15;
  int rOff = (lane < 16) ? 0 : 8;
#pragma unroll
  for (int nt = 0; nt < 4; ++nt)
#pragma unroll
    for (int r = 0; r < 8; ++r)
      ldsP[wave][(rOff + r) * 64 + nt * 16 + col] = f2bf(accp[nt][r] + bvR[nt * 16 + col]);
  __syncthreads();

  v16bf ap[2];
#pragma unroll
  for (int dc = 0; dc < 2; ++dc) {
    int row = lane & 15;
    int koff = (lane < 16) ? 0 : 8;
    const __bf16* r0 = &ldsP[wave][row * 64 + dc * 32 + koff];
    v8bf lo = *(const v8bf*)(r0);
    v8bf hi = *(const v8bf*)(r0 + 16);
    ap[dc] = cat8(lo, hi);
  }

  for (int vt = 0; vt < E_ / 16; ++vt) {
    v16bf bw0 = bfrag_wT_bf16(WvLb, D_, (size_t)(vt * 16), 0, lane);
    v16bf bw1 = bfrag_wT_bf16(WvLb, D_, (size_t)(vt * 16), 32, lane);
    v8f acc = {};
    acc = wmma_bf16(ap[0], bw0, acc);
    acc = wmma_bf16(ap[1], bw1, acc);
    float bias = bvL[vt * 16 + col];
    v8bf o;
#pragma unroll
    for (int r = 0; r < 8; ++r) o[r] = f2bf(acc[r] + bias);
    *(v8bf*)(Vt + (size_t)(vt * 16 + col) * M_ + rowBase + rOff) = o;
  }
}

// ---------------------------------------------------------------------------
// Kernel C: per (b, 16-k-tile) column-wise online softmax stats over q.
// C layout puts lane = column, so each lane owns its column's running
// max/sum; halves merged with a lane-xor-16 shuffle (wave32).
// ---------------------------------------------------------------------------
__global__ void k_stats(const __bf16* __restrict__ Qb, const __bf16* __restrict__ Kb,
                        float* __restrict__ colmax, float* __restrict__ colsum) {
  int lane = threadIdx.x & 31;
  int wave = threadIdx.x >> 5;
  int gw = blockIdx.x * 4 + wave;          // 0 .. B_*(S_/16)-1 = 511
  int b = gw >> 7;
  int kt = gw & 127;
  size_t base = (size_t)b * S_;

  v16bf bk0 = bfrag_wT_bf16(Kb, D_, base + kt * 16, 0, lane);
  v16bf bk1 = bfrag_wT_bf16(Kb, D_, base + kt * 16, 32, lane);

  float rmax = -1.0e30f, rsum = 0.0f;
  for (int qt = 0; qt < S_ / 16; ++qt) {
    v16bf aq0 = afrag_bf16(Qb, D_, base + qt * 16, 0, lane);
    v16bf aq1 = afrag_bf16(Qb, D_, base + qt * 16, 32, lane);
    v8f c = {};
    c = wmma_bf16(aq0, bk0, c);
    c = wmma_bf16(aq1, bk1, c);
#pragma unroll
    for (int r = 0; r < 8; ++r) {
      float s = c[r] * 0.125f;             // 1/sqrt(64)
      float nm = fmaxf(rmax, s);
      rsum = rsum * __expf(rmax - nm) + __expf(s - nm);
      rmax = nm;
    }
  }
  float omax = __shfl_xor(rmax, 16, 32);
  float osum = __shfl_xor(rsum, 16, 32);
  float m = fmaxf(rmax, omax);
  float sm = rsum * __expf(rmax - m) + osum * __expf(omax - m);
  if (lane < 16) {
    colmax[base + kt * 16 + lane] = m;
    colsum[base + kt * 16 + lane] = sm;
  }
}

// ---------------------------------------------------------------------------
// Kernel D: out = attn @ V. One BLOCK (4 waves) per (b, 16-q-tile).
// Per super-iteration of 128 k's: wave w computes the normalized p-tile for
// k-chunk w (4 score WMMAs + exp) into LDS; after the barrier every wave
// consumes all 4 chunks against its private 256-column slice of V
// (16 f32 accumulators). V-fragments loaded in groups of 4 per WMMA group.
// ---------------------------------------------------------------------------
__global__ void k_out(const __bf16* __restrict__ Qb, const __bf16* __restrict__ Kb,
                      const __bf16* __restrict__ Vt,
                      const float* __restrict__ colmax, const float* __restrict__ colsum,
                      float* __restrict__ out) {
  __shared__ __bf16 ldsP[4][16 * 32];      // 4 k-chunks of 32, each 16q x 32k
  int lane = threadIdx.x & 31;
  int wave = threadIdx.x >> 5;
  int gw = blockIdx.x;                     // 0 .. B_*(S_/16)-1 = 511
  int b = gw >> 7;
  int qt = gw & 127;
  size_t base = (size_t)b * S_;
  int col = lane & 15;
  int rOff = (lane < 16) ? 0 : 8;
  int v0 = wave * 256;                     // this wave's 256 output columns
  const __bf16* Vtb = Vt + base;           // shift into this batch's columns

  v16bf aq0 = afrag_bf16(Qb, D_, base + qt * 16, 0, lane);
  v16bf aq1 = afrag_bf16(Qb, D_, base + qt * 16, 32, lane);

  v8f acc[16] = {};
  for (int kb = 0; kb < S_; kb += 128) {   // 16 super-iterations
    {
      int kcLoc = kb + wave * 32;          // this wave's p chunk
#pragma unroll
      for (int sub = 0; sub < 2; ++sub) {
        size_t k0 = base + kcLoc + sub * 16;
        v16bf bk0 = bfrag_wT_bf16(Kb, D_, k0, 0, lane);
        v16bf bk1 = bfrag_wT_bf16(Kb, D_, k0, 32, lane);
        v8f c = {};
        c = wmma_bf16(aq0, bk0, c);
        c = wmma_bf16(aq1, bk1, c);
        float mx = colmax[k0 + col];
        float inv = 1.0f / colsum[k0 + col];
#pragma unroll
        for (int r = 0; r < 8; ++r) {
          float p = __expf(c[r] * 0.125f - mx) * inv;
          ldsP[wave][(rOff + r) * 32 + sub * 16 + col] = f2bf(p);
        }
      }
    }
    __syncthreads();
#pragma unroll
    for (int ch = 0; ch < 4; ++ch) {
      v16bf ap;
      {
        int row = lane & 15;
        int koff = (lane < 16) ? 0 : 8;
        const __bf16* r0 = &ldsP[ch][row * 32 + koff];
        v8bf lo = *(const v8bf*)(r0);
        v8bf hi = *(const v8bf*)(r0 + 16);
        ap = cat8(lo, hi);
      }
      int kk = kb + ch * 32;               // k offset of this chunk (local)
#pragma unroll
      for (int g = 0; g < 4; ++g) {
        v16bf bv[4];
#pragma unroll
        for (int t = 0; t < 4; ++t)
          bv[t] = bfrag_wT_bf16(Vtb, M_, (size_t)(v0 + (g * 4 + t) * 16), kk, lane);
#pragma unroll
        for (int t = 0; t < 4; ++t)
          acc[g * 4 + t] = wmma_bf16(ap, bv[t], acc[g * 4 + t]);
      }
    }
    __syncthreads();
  }
#pragma unroll
  for (int nt = 0; nt < 16; ++nt)
#pragma unroll
    for (int r = 0; r < 8; ++r)
      out[(base + qt * 16 + rOff + r) * E_ + v0 + nt * 16 + col] = acc[nt][r];
}

// ---------------------------------------------------------------------------
extern "C" void kernel_launch(void* const* d_in, const int* in_sizes, int n_in,
                              void* d_out, int out_size, void* d_ws, size_t ws_size,
                              hipStream_t stream) {
  (void)in_sizes; (void)n_in; (void)out_size; (void)ws_size;
  const float* x   = (const float*)d_in[0];
  const float* y   = (const float*)d_in[1];
  const float* Wq  = (const float*)d_in[2];
  const float* bq  = (const float*)d_in[3];
  const float* Wk  = (const float*)d_in[4];
  const float* bk  = (const float*)d_in[5];
  const float* WvR = (const float*)d_in[6];
  const float* bvR = (const float*)d_in[7];
  const float* WvL = (const float*)d_in[8];
  const float* bvL = (const float*)d_in[9];
  float* out = (float*)d_out;

  char* ws = (char*)d_ws;
  __bf16* Qb = (__bf16*)(ws);                                   //  1 MB
  __bf16* Kb = (__bf16*)(ws + ((size_t)1 << 20));               //  1 MB
  __bf16* Vt = (__bf16*)(ws + ((size_t)2 << 20));               // 16 MB (transposed)
  char* tail = ws + ((size_t)18 << 20);
  float*  colmax = (float*)(tail);                              // 32 KB
  float*  colsum = (float*)(tail + (1u << 15));                 // 32 KB
  __bf16* Wqb  = (__bf16*)(tail + (2u << 15));                  // 128 KB each
  __bf16* Wkb  = (__bf16*)(tail + (2u << 15) + (1u << 17));
  __bf16* WvRb = (__bf16*)(tail + (2u << 15) + (2u << 17));
  __bf16* WvLb = (__bf16*)(tail + (2u << 15) + (3u << 17));

  k_cvt_w  <<<(D_ * E_) / 1024, 256, 0, stream>>>(Wq, Wk, WvR, WvL, Wqb, Wkb, WvRb, WvLb);
  k_proj_qk<<<M_ / 64, 128, 0, stream>>>(x, y, Wqb, bq, Wkb, bk, Qb, Kb);
  k_proj_v <<<M_ / 64, 128, 0, stream>>>(y, WvRb, bvR, WvLb, bvL, Vt);
  k_stats  <<<(B_ * (S_ / 16)) / 4, 128, 0, stream>>>(Qb, Kb, colmax, colsum);
  k_out    <<<B_ * (S_ / 16), 128, 0, stream>>>(Qb, Kb, Vt, colmax, colsum, out);
}
